// LancetBlockConfigurable_69398081568962
// MI455X (gfx1250) — matrix operand order repaired and checked by reading, not compiled, not run on hardware
//
#include <hip/hip_runtime.h>

// ---------------------------------------------------------------------------
// MoE block for MI455X (gfx1250, wave32, WMMA, async global->LDS).
//   x:[8,2048,1024] -> LN -> per-expert FFN (1024->4096 GeLU 4096->1024) -> +x
// Gate logits are discarded by the reference -> skipped.
// Pipeline:
//   wconv   : w1,w2 fp32 [e][k][n] -> bf16 transposed [e][n][k]  (one-time,
//             weights then live in the 192MB L2 across all GEMM tiles)
//   ln      : LayerNorm -> bf16 activations
//   ffn1/2  : 128x128x32 tiled GEMM on v_wmma_f32_16x16x32_bf16, with
//             double-buffered LDS fed by global_load_async_to_lds_b128
// ---------------------------------------------------------------------------

#define D_DIM    1024
#define F_DIM    4096
#define E_NUM    8
#define T_PER_MB 4096
#define TOKENS   16384          // 4 micro-batches * 4096 tokens
#define TPE      512            // tokens per expert per micro-batch chunk
#define M_PER_E  2048           // rows per expert (4 chunks of 512)

#define BM 128
#define BN 128
#define BK 32
#define LDS_PAD 40              // elements; 20-dword stride -> conflict-free b128 reads

typedef unsigned short ushort_t;
typedef __attribute__((ext_vector_type(16))) __bf16 v16bf;
typedef __attribute__((ext_vector_type(8)))  float  v8f;
typedef __attribute__((ext_vector_type(8)))  unsigned short us8;

union ABFrag { v16bf v; us8 h[2]; };

// float -> bf16 bits, round-to-nearest-even
__device__ __forceinline__ ushort_t f2bf(float f) {
  union { float f; unsigned int u; } c; c.f = f;
  unsigned int r = c.u + 0x7FFFu + ((c.u >> 16) & 1u);
  return (ushort_t)(r >> 16);
}
__device__ __forceinline__ unsigned pack2bf(float a, float b) {
  return (unsigned)f2bf(a) | ((unsigned)f2bf(b) << 16);
}

// branch-free tanh-form GeLU on native v_exp/v_rcp (no libm divergence)
__device__ __forceinline__ float fast_gelu(float x) {
  const float y = 0.7978845608028654f * x + 0.035677408136300125f * (x * x * x);
  const float t = __builtin_amdgcn_exp2f(-2.885390081777927f * __builtin_fabsf(y));
  float th = (1.0f - t) * __builtin_amdgcn_rcpf(1.0f + t);
  th = __builtin_copysignf(th, y);
  return 0.5f * x * (1.0f + th);
}

// ---------------------------------------------------------------------------
// Kernel 0: weight transpose + fp32->bf16:  wt[e][n][k] = bf16(w[e][k][n])
// grid (K/32, N/32, E), block 256. 32x32 fp32 tile via LDS.
// ---------------------------------------------------------------------------
__global__ __launch_bounds__(256) void wconv_kernel(
    const float* __restrict__ w, ushort_t* __restrict__ wt, int K, int N) {
  __shared__ float tile[32][33];
  const int e  = blockIdx.z;
  const int k0 = blockIdx.x * 32, n0 = blockIdx.y * 32;
  const int tid = threadIdx.x;

  const int tk = tid >> 3, tn = (tid & 7) * 4;
  const float4 f = *(const float4*)(w + ((size_t)e * K + k0 + tk) * N + n0 + tn);
  tile[tk][tn + 0] = f.x; tile[tk][tn + 1] = f.y;
  tile[tk][tn + 2] = f.z; tile[tk][tn + 3] = f.w;
  __syncthreads();

  const int wn = tid >> 3, wk = (tid & 7) * 4;
  uint2 o;
  o.x = pack2bf(tile[wk + 0][wn], tile[wk + 1][wn]);
  o.y = pack2bf(tile[wk + 2][wn], tile[wk + 3][wn]);
  *(uint2*)(wt + ((size_t)e * N + n0 + wn) * K + k0 + wk) = o;
}

// ---------------------------------------------------------------------------
// Kernel 1: LayerNorm (biased var, eps=1e-5) -> bf16 activations.
// ---------------------------------------------------------------------------
__global__ __launch_bounds__(256) void ln_kernel(
    const float* __restrict__ x, const float* __restrict__ gamma,
    const float* __restrict__ beta, ushort_t* __restrict__ xn) {
  __shared__ float red[256];
  const int t   = blockIdx.x;
  const int tid = threadIdx.x;
  const float4 v = ((const float4*)(x + (size_t)t * D_DIM))[tid];

  red[tid] = v.x + v.y + v.z + v.w; __syncthreads();
  #pragma unroll
  for (int off = 128; off > 0; off >>= 1) {
    if (tid < off) red[tid] += red[tid + off];
    __syncthreads();
  }
  const float mu = red[0] * (1.0f / D_DIM);
  __syncthreads();

  const float d0 = v.x - mu, d1 = v.y - mu, d2 = v.z - mu, d3 = v.w - mu;
  red[tid] = d0 * d0 + d1 * d1 + d2 * d2 + d3 * d3; __syncthreads();
  #pragma unroll
  for (int off = 128; off > 0; off >>= 1) {
    if (tid < off) red[tid] += red[tid + off];
    __syncthreads();
  }
  const float rstd = rsqrtf(red[0] * (1.0f / D_DIM) + 1e-5f);

  const int d = tid * 4;
  const float4 g = ((const float4*)gamma)[tid];
  const float4 b = ((const float4*)beta)[tid];
  uint2 pk;
  pk.x = pack2bf(d0 * rstd * g.x + b.x, d1 * rstd * g.y + b.y);
  pk.y = pack2bf(d2 * rstd * g.z + b.z, d3 * rstd * g.w + b.w);
  *(uint2*)(xn + (size_t)t * D_DIM + d) = pk;
}

// ---------------------------------------------------------------------------
// GEMM core: C[128x128] += A[128xK](bf16) * Bt[128xK](bf16, weights^T).
// 8 waves (4x2), wave owns 32x64 = 2x4 WMMA tiles. Double-buffered LDS,
// tiles staged with global_load_async_to_lds_b128 (ASYNCcnt pipeline).
// ---------------------------------------------------------------------------
template<int KTOT>
__device__ __forceinline__ void gemm_tile_async(
    const ushort_t* __restrict__ A,  int lda,   // activations, row-major [128][KTOT]
    const ushort_t* __restrict__ Bt, int ldb,   // weights^T,   row-major [128][KTOT]
    v8f (&acc)[2][4]) {
  __shared__ __align__(16) ushort_t sA[2][BM * LDS_PAD];
  __shared__ __align__(16) ushort_t sB[2][BN * LDS_PAD];

  const int tid  = threadIdx.x;
  const int lane = tid & 31, wid = tid >> 5;
  const int wm = wid & 3, wn = wid >> 2;     // wave: 32 rows x 64 cols
  const int lr = lane & 15, lh = lane >> 4;

  // staging: 2 threads per row, 32B each as 2x b128 async copies
  const int srow = tid >> 1;
  const int skq  = (tid & 1) * 16;           // element offset within row
  const ushort_t* gA = A  + (size_t)srow * lda + skq;
  const ushort_t* gB = Bt + (size_t)srow * ldb + skq;
  const unsigned  lAo = (unsigned)(srow * LDS_PAD + skq) * 2u;
  const unsigned  lBo = lAo;

  auto stage = [&](int buf, int k0) {
    unsigned la = (unsigned)(uintptr_t)(&sA[buf][0]) + lAo;
    unsigned lb = (unsigned)(uintptr_t)(&sB[buf][0]) + lBo;
    const ushort_t* pa = gA + k0;
    const ushort_t* pb = gB + k0;
    asm volatile(
        "global_load_async_to_lds_b128 %0, %2, off\n\t"
        "global_load_async_to_lds_b128 %0, %2, off offset:16\n\t"
        "global_load_async_to_lds_b128 %1, %3, off\n\t"
        "global_load_async_to_lds_b128 %1, %3, off offset:16"
        :: "v"(la), "v"(lb), "v"(pa), "v"(pb) : "memory");
  };

  stage(0, 0);
  asm volatile("s_wait_asynccnt 0x0" ::: "memory");
  __syncthreads();

  int p = 0;
  for (int k0 = 0; k0 < KTOT; k0 += BK) {
    if (k0 + BK < KTOT) stage(p ^ 1, k0 + BK);

    ABFrag a[2], b[4];
    #pragma unroll
    for (int mt = 0; mt < 2; ++mt) {
      // A lane (low half): K 0..7 & 16..23 ; (high half): K 8..15 & 24..31
      const ushort_t* ap = &sA[p][(wm * 32 + mt * 16 + lr) * LDS_PAD + lh * 8];
      a[mt].h[0] = *(const us8*)ap;
      a[mt].h[1] = *(const us8*)(ap + 16);
    }
    #pragma unroll
    for (int nt = 0; nt < 4; ++nt) {
      // B lane (low half): K 0..7 & 16..23 ; (high half): K 8..15 & 24..31
      const ushort_t* bp = &sB[p][(wn * 64 + nt * 16 + lr) * LDS_PAD + lh * 8];
      b[nt].h[0] = *(const us8*)bp;
      b[nt].h[1] = *(const us8*)(bp + 16);
    }
    #pragma unroll
    for (int mt = 0; mt < 2; ++mt)
      #pragma unroll
      for (int nt = 0; nt < 4; ++nt)
        acc[mt][nt] = __builtin_amdgcn_wmma_f32_16x16x32_bf16(
            false, a[mt].v, false, b[nt].v, (short)0, acc[mt][nt], false, false);

    asm volatile("s_wait_asynccnt 0x0" ::: "memory");
    __syncthreads();
    p ^= 1;
  }
}

// expert-local row -> global flat token row (512-token round-robin chunks)
__device__ __forceinline__ int expert_row_base(int e, int local0) {
  const int mb = local0 >> 9;
  return mb * T_PER_MB + e * TPE + (local0 & 511);
}

// ---------------------------------------------------------------------------
// Kernel 2: H = gelu(Xn_e * W1[e] + b1[e]) -> bf16. grid=(32,16,8)
// ---------------------------------------------------------------------------
__global__ __launch_bounds__(256) void ffn1_kernel(
    const ushort_t* __restrict__ xn, const ushort_t* __restrict__ wt1,
    const float* __restrict__ b1, ushort_t* __restrict__ hbuf) {
  const int e = blockIdx.z, by = blockIdx.y, bx = blockIdx.x;
  const int local0 = by * BM;
  const int g0 = expert_row_base(e, local0);

  const ushort_t* A  = xn  + (size_t)g0 * D_DIM;
  const ushort_t* Bt = wt1 + ((size_t)e * F_DIM + bx * BN) * D_DIM;

  v8f acc[2][4] = {};
  gemm_tile_async<D_DIM>(A, D_DIM, Bt, D_DIM, acc);

  const int tid = threadIdx.x, lane = tid & 31, wid = tid >> 5;
  const int wm = wid & 3, wn = wid >> 2, lr = lane & 15, lh = lane >> 4;
  ushort_t* H = hbuf + ((size_t)e * M_PER_E + local0) * F_DIM + bx * BN;

  #pragma unroll
  for (int mt = 0; mt < 2; ++mt)
    #pragma unroll
    for (int nt = 0; nt < 4; ++nt) {
      const int col = wn * 64 + nt * 16 + lr;
      const float bias = b1[(size_t)e * F_DIM + bx * BN + col];
      #pragma unroll
      for (int j = 0; j < 8; ++j) {
        const int row = wm * 32 + mt * 16 + lh * 8 + j;
        H[(size_t)row * F_DIM + col] = f2bf(fast_gelu(acc[mt][nt][j] + bias));
      }
    }
}

// ---------------------------------------------------------------------------
// Kernel 3: out = x + (H * W2[e] + b2[e]) fp32. grid=(8,16,8)
// ---------------------------------------------------------------------------
__global__ __launch_bounds__(256) void ffn2_kernel(
    const ushort_t* __restrict__ hbuf, const ushort_t* __restrict__ wt2,
    const float* __restrict__ b2, const float* __restrict__ x,
    float* __restrict__ out) {
  const int e = blockIdx.z, by = blockIdx.y, bx = blockIdx.x;
  const int local0 = by * BM;
  const int g0 = expert_row_base(e, local0);

  const ushort_t* A  = hbuf + ((size_t)e * M_PER_E + local0) * F_DIM;
  const ushort_t* Bt = wt2  + ((size_t)e * D_DIM + bx * BN) * F_DIM;

  v8f acc[2][4] = {};
  gemm_tile_async<F_DIM>(A, F_DIM, Bt, F_DIM, acc);

  const int tid = threadIdx.x, lane = tid & 31, wid = tid >> 5;
  const int wm = wid & 3, wn = wid >> 2, lr = lane & 15, lh = lane >> 4;

  #pragma unroll
  for (int mt = 0; mt < 2; ++mt)
    #pragma unroll
    for (int nt = 0; nt < 4; ++nt) {
      const int col = bx * BN + wn * 64 + nt * 16 + lr;
      const float bias = b2[(size_t)e * D_DIM + col];
      #pragma unroll
      for (int j = 0; j < 8; ++j) {
        const int grow = g0 + wm * 32 + mt * 16 + lh * 8 + j;
        out[(size_t)grow * D_DIM + col] =
            acc[mt][nt][j] + bias + x[(size_t)grow * D_DIM + col];
      }
    }
}

// ---------------------------------------------------------------------------
// Launch. ws layout (bf16 elements):
//   xn [16384*1024] | H [8*2048*4096] | wt1 [8*4096*1024] | wt2 [8*1024*4096]
//   = 33.5MB + 134MB + 67MB + 67MB ~= 302 MB scratch.
// ---------------------------------------------------------------------------
extern "C" void kernel_launch(void* const* d_in, const int* in_sizes, int n_in,
                              void* d_out, int out_size, void* d_ws, size_t ws_size,
                              hipStream_t stream) {
  const float* x     = (const float*)d_in[0];
  const float* gamma = (const float*)d_in[1];
  const float* beta  = (const float*)d_in[2];
  // d_in[3] = gate_w : reference discards gate output -> skipped
  const float* w1    = (const float*)d_in[4];
  const float* b1    = (const float*)d_in[5];
  const float* w2    = (const float*)d_in[6];
  const float* b2    = (const float*)d_in[7];
  float* out = (float*)d_out;

  ushort_t* xn  = (ushort_t*)d_ws;
  ushort_t* hb  = xn  + (size_t)TOKENS * D_DIM;
  ushort_t* wt1 = hb  + (size_t)E_NUM * M_PER_E * F_DIM;
  ushort_t* wt2 = wt1 + (size_t)E_NUM * D_DIM * F_DIM;

  wconv_kernel<<<dim3(D_DIM / 32, F_DIM / 32, E_NUM), 256, 0, stream>>>(w1, wt1, D_DIM, F_DIM);
  wconv_kernel<<<dim3(F_DIM / 32, D_DIM / 32, E_NUM), 256, 0, stream>>>(w2, wt2, F_DIM, D_DIM);
  ln_kernel<<<TOKENS, 256, 0, stream>>>(x, gamma, beta, xn);
  ffn1_kernel<<<dim3(F_DIM / BN, M_PER_E / BM, E_NUM), 256, 0, stream>>>(xn, wt1, b1, hb);
  ffn2_kernel<<<dim3(D_DIM / BN, M_PER_E / BM, E_NUM), 256, 0, stream>>>(hb, wt2, b2, x, out);
}
